// AdaptiveLiquidNetwork_16544214024866
// MI455X (gfx1250) — matrix operand order repaired and checked by reading, not compiled, not run on hardware
//
#include <hip/hip_runtime.h>

#define B_SZ  16384
#define D_IN  1024
#define UNITS 512
#define D_OUT 512

typedef __attribute__((ext_vector_type(16))) _Float16 v16h;
typedef __attribute__((ext_vector_type(8)))  _Float16 v8h;
typedef __attribute__((ext_vector_type(8)))  float    v8f;
typedef __attribute__((ext_vector_type(4)))  float    v4f;

union Frag16 { v16h v; v8h h[2]; };

#define BM 128
#define BN 64
#define BK 32
#define LDSROW (BK + 8)   // +16B pad per row to break bank phase

enum { EP_TANH = 0, EP_LIQUID = 1, EP_BIAS = 2 };

// --- CDNA5 helpers ---------------------------------------------------------

// Async copy global -> LDS (16B per lane), tracked on ASYNCcnt.
// LDS address = low 32 bits of the generic shared pointer (flat->LDS per ISA 10.2).
__device__ __forceinline__ void async_copy_b128(const void* gptr, void* lptr) {
  unsigned lds          = (unsigned)(size_t)lptr;
  unsigned long long ga = (unsigned long long)gptr;
  asm volatile("global_load_async_to_lds_b128 %0, %1, off"
               :: "v"(lds), "v"(ga) : "memory");
}
__device__ __forceinline__ void wait_async0() {
  asm volatile("s_wait_asynccnt 0x0" ::: "memory");
}

__device__ __forceinline__ float fast_tanh(float x) {
#if __has_builtin(__builtin_amdgcn_tanhf)
  return __builtin_amdgcn_tanhf(x);          // V_TANH_F32 (gfx1250 TRANS op)
#else
  float t = __expf(2.0f * x);
  return 1.0f - 2.0f / (t + 1.0f);
#endif
}
__device__ __forceinline__ float fast_sigmoid(float x) {
  return 1.0f / (1.0f + __expf(-x));
}

// ---------------------------------------------------------------------------
// Fused WMMA GEMM:  D[M,N] = epilogue( A[M,K] @ Bw[N,K]^T + bias[N] )
// A is f32 (converted inline, register-staged) or f16 (async-copied to LDS);
// Bw is pre-converted f16, row = N, K-contiguous (always async-copied).
// Block: 256 thr = 8 wave32. Block tile 128x64, wave tile 32x32 (2x2 WMMA 16x16x32).
// ---------------------------------------------------------------------------
template<int EPI, bool AF32>
__global__ __launch_bounds__(256, 2) void lnn_gemm_kernel(
    const float* __restrict__ Af, const _Float16* __restrict__ Ah,
    const _Float16* __restrict__ Bw,
    const float* __restrict__ bias, const float* __restrict__ tau,
    const float* __restrict__ states,
    float* __restrict__ outF, _Float16* __restrict__ outH,
    int M, int N, int K)
{
  (void)M;
  __shared__ __align__(16) _Float16 As[2][BM][LDSROW];
  __shared__ __align__(16) _Float16 Bs[2][BN][LDSROW];

  const int t    = threadIdx.x;
  const int lane = t & 31;
  const int wave = t >> 5;        // 0..7
  const int wm   = wave & 3;      // wave row 0..3  -> 32 rows each
  const int wn   = wave >> 2;     // wave col 0..1  -> 32 cols each
  const int half = lane >> 4;     // hi/lo 16-lane group
  const int l16  = lane & 15;

  const int rowBase = blockIdx.y * BM;
  const int colBase = blockIdx.x * BN;

  // staging geometry
  const int ar = t >> 1;          // A: 128 rows, 2 threads/row (16 halves each)
  const int ac = (t & 1) * 16;
  const int br = t >> 2;          // B: 64 rows, 4 threads/row (8 halves each)
  const int bc = (t & 3) * 8;

  // f32-A register staging (GEMM1 only: fp32 x converted inline)
  v8h sA0, sA1;

  auto issue_async = [&](int kc, int buf) {
    const int kof = kc * BK;
    if (!AF32) {
      const _Float16* ga = Ah + (size_t)(rowBase + ar) * K + kof + ac;
      async_copy_b128(ga,     &As[buf][ar][ac]);
      async_copy_b128(ga + 8, &As[buf][ar][ac + 8]);
    }
    const _Float16* gb = Bw + (size_t)(colBase + br) * K + kof + bc;
    async_copy_b128(gb, &Bs[buf][br][bc]);
  };
  auto stageA_load = [&](int kc) {   // AF32 only
    const float* p = Af + (size_t)(rowBase + ar) * K + kc * BK + ac;
    v4f q0 = *(const v4f*)(p + 0);
    v4f q1 = *(const v4f*)(p + 4);
    v4f q2 = *(const v4f*)(p + 8);
    v4f q3 = *(const v4f*)(p + 12);
    #pragma unroll
    for (int i = 0; i < 4; ++i) {
      sA0[i]     = (_Float16)q0[i];
      sA0[4 + i] = (_Float16)q1[i];
      sA1[i]     = (_Float16)q2[i];
      sA1[4 + i] = (_Float16)q3[i];
    }
  };
  auto stageA_write = [&](int buf) { // AF32 only
    *(v8h*)&As[buf][ar][ac]     = sA0;
    *(v8h*)&As[buf][ar][ac + 8] = sA1;
  };

  v8f acc[2][2] = {};

  auto compute = [&](int buf) {
    // WMMA A layout (16x32 f16): lane L row=L%16, halves[0..7]=K half*8..+7,
    // halves[8..15]=K 16+half*8..+7  -> two contiguous 16B LDS reads.
    Frag16 fa[2], fb[2];
    #pragma unroll
    for (int mi = 0; mi < 2; ++mi) {
      const _Float16* ap = &As[buf][wm * 32 + mi * 16 + l16][0];
      fa[mi].h[0] = *(const v8h*)(ap + half * 8);
      fa[mi].h[1] = *(const v8h*)(ap + 16 + half * 8);
    }
    // WMMA B layout (32x16 f16): lane L col=L%16, halves[0..7]=K half*8..+7, halves[8..15]=K 16+half*8..+7
    #pragma unroll
    for (int ni = 0; ni < 2; ++ni) {
      const _Float16* bp = &Bs[buf][wn * 32 + ni * 16 + l16][0];
      fb[ni].h[0] = *(const v8h*)(bp + half * 8);
      fb[ni].h[1] = *(const v8h*)(bp + 16 + half * 8);
    }
    #pragma unroll
    for (int mi = 0; mi < 2; ++mi)
      #pragma unroll
      for (int ni = 0; ni < 2; ++ni)
        acc[mi][ni] = __builtin_amdgcn_wmma_f32_16x16x32_f16(
            false, fa[mi].v, false, fb[ni].v, (short)0, acc[mi][ni], false, false);
  };

  const int nk = K / BK;

  // prologue: fill buffer 0
  issue_async(0, 0);
  if (AF32) { stageA_load(0); stageA_write(0); }

  // steady state: branch-free; next chunk staged while WMMAs run
  for (int kc = 0; kc < nk - 1; ++kc) {
    wait_async0();
    __syncthreads();                 // buf[kc&1] ready; buf[kc&1 ^ 1] free
    issue_async(kc + 1, (kc + 1) & 1);
    if (AF32) stageA_load(kc + 1);
    compute(kc & 1);
    if (AF32) stageA_write((kc + 1) & 1);
  }
  wait_async0();
  __syncthreads();
  compute((nk - 1) & 1);

  // Epilogue. C/D layout: lane L, VGPR i -> (M = i + 8*half, N = l16)
  #pragma unroll
  for (int mi = 0; mi < 2; ++mi) {
    #pragma unroll
    for (int ni = 0; ni < 2; ++ni) {
      const int row0 = rowBase + wm * 32 + mi * 16 + half * 8;
      const int col  = colBase + wn * 32 + ni * 16 + l16;
      const float bv = bias[col];
      if (EPI == EP_TANH) {
        #pragma unroll
        for (int i = 0; i < 8; ++i) {
          float v = fast_tanh(acc[mi][ni][i] + bv);
          outH[(size_t)(row0 + i) * N + col] = (_Float16)v;
        }
      } else if (EPI == EP_LIQUID) {
        const float dec = __expf(-0.1f / tau[col]);
        #pragma unroll
        for (int i = 0; i < 8; ++i) {
          float a  = fast_sigmoid(acc[mi][ni][i] + bv);
          float st = states[(size_t)(row0 + i) * N + col];
          float ns = a + (st - a) * dec;
          outF[(size_t)(row0 + i) * N + col] = ns;            // fp32 new_states
          outH[(size_t)(row0 + i) * N + col] = (_Float16)ns;  // f16 for GEMM3
        }
      } else {
        #pragma unroll
        for (int i = 0; i < 8; ++i)
          outF[(size_t)(row0 + i) * N + col] = acc[mi][ni][i] + bv;
      }
    }
  }
}

// Weight prep: fp32 -> f16 copy, and fp32 -> f16 transpose (for sensory_w,
// used as projected @ sensory_w, so we need [N,K] = sensory_w^T)
__global__ void cvt_f32_to_f16_kernel(const float* __restrict__ src,
                                      _Float16* __restrict__ dst, int n) {
  int i = blockIdx.x * blockDim.x + threadIdx.x;
  if (i < n) dst[i] = (_Float16)src[i];
}

__global__ void transpose_cvt_kernel(const float* __restrict__ src,
                                     _Float16* __restrict__ dst, int dim) {
  __shared__ float tile[32][33];
  int x = blockIdx.x * 32 + threadIdx.x;  // col (n)
  int y = blockIdx.y * 32 + threadIdx.y;  // row (k)
  tile[threadIdx.y][threadIdx.x] = src[(size_t)y * dim + x];
  __syncthreads();
  int ox = blockIdx.y * 32 + threadIdx.x; // k
  int oy = blockIdx.x * 32 + threadIdx.y; // n
  dst[(size_t)oy * dim + ox] = (_Float16)tile[threadIdx.x][threadIdx.y];
}

extern "C" void kernel_launch(void* const* d_in, const int* in_sizes, int n_in,
                              void* d_out, int out_size, void* d_ws, size_t ws_size,
                              hipStream_t stream) {
  (void)in_sizes; (void)n_in; (void)out_size; (void)ws_size;
  const float* x        = (const float*)d_in[0];
  const float* w_in     = (const float*)d_in[1];
  const float* b_in     = (const float*)d_in[2];
  const float* sens_w   = (const float*)d_in[3];
  const float* sens_sig = (const float*)d_in[4];
  const float* tau      = (const float*)d_in[5];
  const float* ro_w     = (const float*)d_in[6];
  const float* ro_b     = (const float*)d_in[7];
  const float* states   = (const float*)d_in[8];

  float* out    = (float*)d_out;                 // [B, D_OUT]
  float* out_ns = out + (size_t)B_SZ * D_OUT;    // [B, UNITS]

  // workspace layout (f16): w_in | sensory^T | readout | projected | new_states
  _Float16* w_in_h = (_Float16*)d_ws;
  _Float16* sens_h = w_in_h + (size_t)UNITS * D_IN;
  _Float16* ro_h   = sens_h + (size_t)UNITS * UNITS;
  _Float16* proj_h = ro_h   + (size_t)D_OUT * UNITS;
  _Float16* ns_h   = proj_h + (size_t)B_SZ * UNITS;

  cvt_f32_to_f16_kernel<<<(UNITS * D_IN) / 256, 256, 0, stream>>>(w_in, w_in_h, UNITS * D_IN);
  cvt_f32_to_f16_kernel<<<(D_OUT * UNITS) / 256, 256, 0, stream>>>(ro_w, ro_h, D_OUT * UNITS);
  transpose_cvt_kernel<<<dim3(UNITS / 32, UNITS / 32), dim3(32, 32), 0, stream>>>(sens_w, sens_h, UNITS);

  dim3 blk(256);
  // GEMM1: projected = tanh(x @ w_in^T + b_in)  [f16 out]
  lnn_gemm_kernel<EP_TANH, true><<<dim3(UNITS / BN, B_SZ / BM), blk, 0, stream>>>(
      x, nullptr, w_in_h, b_in, nullptr, nullptr, nullptr, proj_h, B_SZ, UNITS, D_IN);
  // GEMM2: A = sigmoid(projected @ sensory_w + sigma); ns = A + (states-A)*exp(-t/tau)
  lnn_gemm_kernel<EP_LIQUID, false><<<dim3(UNITS / BN, B_SZ / BM), blk, 0, stream>>>(
      nullptr, proj_h, sens_h, sens_sig, tau, states, out_ns, ns_h, B_SZ, UNITS, UNITS);
  // GEMM3: out = ns @ readout_w^T + readout_b  [fp32 out]
  lnn_gemm_kernel<EP_BIAS, false><<<dim3(D_OUT / BN, B_SZ / BM), blk, 0, stream>>>(
      nullptr, ns_h, ro_h, ro_b, nullptr, nullptr, out, nullptr, B_SZ, D_OUT, UNITS);
}
